// AwkwardRNN_44401371906313
// MI455X (gfx1250) — compile-verified, matrix operand order checked by reading, not run on hardware
//
#include <hip/hip_runtime.h>
#include <cstdint>

#define H2 2048
#define G  8192   // 4*H2
#define TT 2048
#define LL 4
#define SCAN_BLOCKS 64

typedef __bf16 bf16x8  __attribute__((ext_vector_type(8)));
typedef __bf16 bf16x16 __attribute__((ext_vector_type(16)));
typedef float  f32x8   __attribute__((ext_vector_type(8)));

// ---------- helpers ----------
__device__ __forceinline__ unsigned short f2bf(float x) {
    union { float f; unsigned u; } v; v.f = x;
    unsigned r = v.u + 0x7FFFu + ((v.u >> 16) & 1u);   // round-to-nearest-even
    return (unsigned short)(r >> 16);
}
__device__ __forceinline__ unsigned pk2(float a, float b) {
    return (unsigned)f2bf(a) | ((unsigned)f2bf(b) << 16);
}
__device__ __forceinline__ float bflo(unsigned u) { return __uint_as_float(u << 16); }
__device__ __forceinline__ float bfhi(unsigned u) { return __uint_as_float(u & 0xFFFF0000u); }
__device__ __forceinline__ float sigf(float x)    { return 1.0f / (1.0f + __expf(-x)); }

__device__ __forceinline__ bf16x16 ldfrag(const unsigned short* p) {
    bf16x8 lo = *(const bf16x8*)p;          // K chunk +0  (8 bf16, 16B)
    bf16x8 hi = *(const bf16x8*)(p + 16);   // K chunk +16 elements
    bf16x16 r;
#pragma unroll
    for (int i = 0; i < 8; ++i) { r[i] = lo[i]; r[i + 8] = hi[i]; }
    return r;
}

// device-scope grid barrier for the persistent scan kernel
__device__ __forceinline__ void grid_barrier(unsigned int* bar, unsigned int& gen) {
    __syncthreads();
    if (threadIdx.x == 0) {
        unsigned arrived = __hip_atomic_fetch_add(&bar[0], 1u, __ATOMIC_ACQ_REL,
                                                  __HIP_MEMORY_SCOPE_AGENT);
        if (arrived == gridDim.x - 1) {
            __hip_atomic_store(&bar[0], 0u, __ATOMIC_RELAXED, __HIP_MEMORY_SCOPE_AGENT);
            __hip_atomic_fetch_add(&bar[1], 1u, __ATOMIC_RELEASE, __HIP_MEMORY_SCOPE_AGENT);
        } else {
            while (__hip_atomic_load(&bar[1], __ATOMIC_ACQUIRE,
                                     __HIP_MEMORY_SCOPE_AGENT) <= gen)
                __builtin_amdgcn_s_sleep(2);
        }
        gen++;
    }
    __syncthreads();
}

// ---------- kernels ----------

// layer 0: xw[t][j] = event[t]*w_ih0[j] + b_ih[j] + b_hh[j]   (rank-1, no GEMM)
__global__ __launch_bounds__(256) void k_xw0(const float* __restrict__ event,
                                             const float* __restrict__ w0,
                                             const float* __restrict__ bi,
                                             const float* __restrict__ bh,
                                             float* __restrict__ xw) {
    int idx = blockIdx.x * 256 + threadIdx.x;   // < TT*G
    int t = idx >> 13;                          // /8192
    int j = idx & (G - 1);
    xw[idx] = fmaf(event[t], w0[j], bi[j] + bh[j]);
}

// w_hh (8192x2048 fp32, gate-major rows) -> bf16 packed 8-wide along K:
// wq4[(k8*4 + g)*H2 + j] = uint4{ k=8k8..8k8+7 of row (g*H2+j) as bf16 pairs }
__global__ __launch_bounds__(256) void k_conv_whh(const float* __restrict__ w,
                                                  uint4* __restrict__ wq4) {
    int idx = blockIdx.x * 256 + threadIdx.x;   // < (H2/8)*4*H2 = 2M
    int j  = idx & (H2 - 1);
    int g  = (idx >> 11) & 3;
    int k8 = idx >> 13;
    const float* row = w + (size_t)(g * H2 + j) * H2 + 8 * k8;
    uint4 o;
    o.x = pk2(row[0], row[1]);
    o.y = pk2(row[2], row[3]);
    o.z = pk2(row[4], row[5]);
    o.w = pk2(row[6], row[7]);
    wq4[idx] = o;
}

// w_ih (8192x2048 fp32) -> bf16, same row-major layout (GEMM B operand)
__global__ __launch_bounds__(256) void k_conv_wib(const float* __restrict__ w,
                                                  unsigned short* __restrict__ wb) {
    int idx = blockIdx.x * 256 + threadIdx.x;   // < G*H2
    wb[idx] = f2bf(w[idx]);
}

__global__ void k_bar_init(unsigned int* bar) { bar[0] = 0u; bar[1] = 0u; }

// xw = X(T x H2, bf16) @ W^T (W: G x H2, bf16) + (b_ih + b_hh)
// wave tile: 16(M) x 64(N); block: 8 waves -> 32(M) x 256(N)
// software-pipelined: loads for kb+32 issued before WMMAs on kb fragments.
__global__ __launch_bounds__(256) void k_gemm(const unsigned short* __restrict__ X,
                                              const unsigned short* __restrict__ Wb,
                                              const float* __restrict__ bih,
                                              const float* __restrict__ bhh,
                                              float* __restrict__ xw) {
    const int lane = threadIdx.x & 31;
    const int wave = threadIdx.x >> 5;
    const int half = lane >> 4;
    const int l16  = lane & 15;
    const int m0 = blockIdx.y * 32 + (wave & 1) * 16;
    const int n0 = blockIdx.x * 256 + (wave >> 1) * 64;

    const unsigned short* ap  = X  + (size_t)(m0 + l16) * H2 + half * 8;
    const unsigned short* bp0 = Wb + (size_t)(n0 +  0 + l16) * H2 + half * 8;
    const unsigned short* bp1 = bp0 + 16 * H2;
    const unsigned short* bp2 = bp0 + 32 * H2;
    const unsigned short* bp3 = bp0 + 48 * H2;

    f32x8 acc0 = {}, acc1 = {}, acc2 = {}, acc3 = {};

    bf16x16 A  = ldfrag(ap);
    bf16x16 B0 = ldfrag(bp0);
    bf16x16 B1 = ldfrag(bp1);
    bf16x16 B2 = ldfrag(bp2);
    bf16x16 B3 = ldfrag(bp3);

    for (int kb = 32; kb < H2; kb += 32) {
        bf16x16 An  = ldfrag(ap  + kb);
        bf16x16 Bn0 = ldfrag(bp0 + kb);
        bf16x16 Bn1 = ldfrag(bp1 + kb);
        bf16x16 Bn2 = ldfrag(bp2 + kb);
        bf16x16 Bn3 = ldfrag(bp3 + kb);
        acc0 = __builtin_amdgcn_wmma_f32_16x16x32_bf16(false, A, false, B0, (short)0, acc0, false, false);
        acc1 = __builtin_amdgcn_wmma_f32_16x16x32_bf16(false, A, false, B1, (short)0, acc1, false, false);
        acc2 = __builtin_amdgcn_wmma_f32_16x16x32_bf16(false, A, false, B2, (short)0, acc2, false, false);
        acc3 = __builtin_amdgcn_wmma_f32_16x16x32_bf16(false, A, false, B3, (short)0, acc3, false, false);
        A = An; B0 = Bn0; B1 = Bn1; B2 = Bn2; B3 = Bn3;
    }
    acc0 = __builtin_amdgcn_wmma_f32_16x16x32_bf16(false, A, false, B0, (short)0, acc0, false, false);
    acc1 = __builtin_amdgcn_wmma_f32_16x16x32_bf16(false, A, false, B1, (short)0, acc1, false, false);
    acc2 = __builtin_amdgcn_wmma_f32_16x16x32_bf16(false, A, false, B2, (short)0, acc2, false, false);
    acc3 = __builtin_amdgcn_wmma_f32_16x16x32_bf16(false, A, false, B3, (short)0, acc3, false, false);

    const float b0 = bih[n0 +  0 + l16] + bhh[n0 +  0 + l16];
    const float b1 = bih[n0 + 16 + l16] + bhh[n0 + 16 + l16];
    const float b2 = bih[n0 + 32 + l16] + bhh[n0 + 32 + l16];
    const float b3 = bih[n0 + 48 + l16] + bhh[n0 + 48 + l16];

    const int rbase = m0 + half * 8;   // D layout: vgpr r -> M = r + 8*half, N = l16
#pragma unroll
    for (int r = 0; r < 8; ++r) {
        float* orow = xw + (size_t)(rbase + r) * G + n0;
        orow[ 0 + l16] = acc0[r] + b0;
        orow[16 + l16] = acc1[r] + b1;
        orow[32 + l16] = acc2[r] + b2;
        orow[48 + l16] = acc3[r] + b3;
    }
}

// persistent LSTM scan: 64 blocks x 128 threads. Block owns 32 hidden indices;
// thread (g, jj) computes the full 2048-K dot for gate row g*H2 + j.
// Gate combine via LDS inside the block; cell state c lives in g==0 registers.
// One device-scope barrier per time step.
__global__ __launch_bounds__(128) void k_scan(const float* __restrict__ xw,
                                              const uint4* __restrict__ wq4,
                                              float* __restrict__ hg,
                                              unsigned short* __restrict__ hseq,
                                              unsigned int* __restrict__ bar) {
    __shared__ float hs[H2];
    __shared__ float gacc[4][32];
    const int g  = threadIdx.x >> 5;            // gate 0..3 (one wave per gate)
    const int jj = threadIdx.x & 31;
    const int j  = blockIdx.x * 32 + jj;        // hidden index 0..2047
    unsigned gen = 0;
    float c = 0.f;

    if (g == 0) hg[j] = 0.f;                    // h_0 = 0
    __threadfence();
    grid_barrier(bar, gen);

    const uint4* wp = wq4 + (size_t)g * H2 + j; // element (k8*4+g)*H2 + j

    for (int t = 0; t < TT; ++t) {
        for (int i = threadIdx.x; i < H2; i += 128) hs[i] = hg[i];
        __syncthreads();

        float a = xw[(size_t)t * G + (size_t)g * H2 + j];
        if (t + 1 < TT)
            __builtin_prefetch(xw + (size_t)(t + 1) * G + (size_t)g * H2 + j, 0, 0);

        const float4* h4 = (const float4*)hs;
#pragma unroll 4
        for (int k8 = 0; k8 < H2 / 8; ++k8) {
            const uint4  u  = wp[(size_t)k8 * (4 * H2)];
            const float4 hA = h4[2 * k8];
            const float4 hB = h4[2 * k8 + 1];
            a = fmaf(bflo(u.x), hA.x, a); a = fmaf(bfhi(u.x), hA.y, a);
            a = fmaf(bflo(u.y), hA.z, a); a = fmaf(bfhi(u.y), hA.w, a);
            a = fmaf(bflo(u.z), hB.x, a); a = fmaf(bfhi(u.z), hB.y, a);
            a = fmaf(bflo(u.w), hB.z, a); a = fmaf(bfhi(u.w), hB.w, a);
        }
        gacc[g][jj] = a;
        __syncthreads();

        if (g == 0) {
            float i_ = sigf(gacc[0][jj]);
            float f_ = sigf(gacc[1][jj]);
            float g_ = tanhf(gacc[2][jj]);
            float o_ = sigf(gacc[3][jj]);
            c = fmaf(f_, c, i_ * g_);
            float h = o_ * tanhf(c);
            hg[j] = h;
            hseq[(size_t)t * H2 + j] = f2bf(h);
        }
        __threadfence();
        grid_barrier(bar, gen);
    }
}

// logits = h @ w_out^T + b_out ; out = log_softmax(logits)  (2 classes)
__global__ __launch_bounds__(256) void k_final(const float* __restrict__ h,
                                               const float* __restrict__ w_out,
                                               const float* __restrict__ b_out,
                                               float* __restrict__ out) {
    __shared__ float s0[256], s1[256];
    float a = 0.f, b = 0.f;
    for (int i = threadIdx.x; i < H2; i += 256) {
        float hv = h[i];
        a = fmaf(hv, w_out[i], a);
        b = fmaf(hv, w_out[H2 + i], b);
    }
    s0[threadIdx.x] = a; s1[threadIdx.x] = b;
    __syncthreads();
    for (int st = 128; st > 0; st >>= 1) {
        if (threadIdx.x < st) { s0[threadIdx.x] += s0[threadIdx.x + st];
                                s1[threadIdx.x] += s1[threadIdx.x + st]; }
        __syncthreads();
    }
    if (threadIdx.x == 0) {
        float l0 = s0[0] + b_out[0], l1 = s1[0] + b_out[1];
        float m = fmaxf(l0, l1);
        float lse = m + logf(expf(l0 - m) + expf(l1 - m));
        out[0] = l0 - lse; out[1] = l1 - lse;
    }
}

// ---------- host ----------
extern "C" void kernel_launch(void* const* d_in, const int* in_sizes, int n_in,
                              void* d_out, int out_size, void* d_ws, size_t ws_size,
                              hipStream_t stream) {
    (void)in_sizes; (void)n_in; (void)out_size; (void)ws_size;
    const float* event = (const float*)d_in[0];
    const float* w_ih0 = (const float*)d_in[1];
    const float* w_ih  = (const float*)d_in[2];   // (L-1, G, H2)
    const float* w_hh  = (const float*)d_in[3];   // (L, G, H2)
    const float* b_ih  = (const float*)d_in[4];   // (L, G)
    const float* b_hh  = (const float*)d_in[5];   // (L, G)
    const float* w_out = (const float*)d_in[6];
    const float* b_out = (const float*)d_in[7];

    char* base = (char*)d_ws;
    size_t off = 0;
    auto al = [](size_t x) { return (x + 255) & ~(size_t)255; };
    float*          xw   = (float*)(base + off);          off = al(off + (size_t)TT * G * 4);
    uint4*          wq4  = (uint4*)(base + off);          off = al(off + (size_t)G * H2 * 2);
    unsigned short* wib  = (unsigned short*)(base + off); off = al(off + (size_t)G * H2 * 2);
    unsigned short* hseq = (unsigned short*)(base + off); off = al(off + (size_t)TT * H2 * 2);
    float*          hg   = (float*)(base + off);          off = al(off + (size_t)H2 * 4);
    unsigned int*   bar  = (unsigned int*)(base + off);   off = al(off + 256);

    dim3 b256(256);

    // layer 0 (rank-1 input path)
    k_xw0<<<(TT * G) / 256, b256, 0, stream>>>(event, w_ih0, b_ih, b_hh, xw);
    k_conv_whh<<<(G * H2 / 8) / 256, b256, 0, stream>>>(w_hh, wq4);
    k_bar_init<<<1, 1, 0, stream>>>(bar);
    k_scan<<<SCAN_BLOCKS, 128, 0, stream>>>(xw, wq4, hg, hseq, bar);

    for (int l = 1; l < LL; ++l) {
        k_conv_wib<<<(G * H2) / 256, b256, 0, stream>>>(w_ih + (size_t)(l - 1) * G * H2, wib);
        k_gemm<<<dim3(G / 256, TT / 32), b256, 0, stream>>>(hseq, wib,
                                                            b_ih + (size_t)l * G,
                                                            b_hh + (size_t)l * G, xw);
        k_conv_whh<<<(G * H2 / 8) / 256, b256, 0, stream>>>(w_hh + (size_t)l * G * H2, wq4);
        k_bar_init<<<1, 1, 0, stream>>>(bar);
        k_scan<<<SCAN_BLOCKS, 128, 0, stream>>>(xw, wq4, hg, hseq, bar);
    }

    k_final<<<1, b256, 0, stream>>>(hg, w_out, b_out, (float*)d_out);
}
